// VulnerabilityGNN_46926812676505
// MI455X (gfx1250) — matrix operand order repaired and checked by reading, not compiled, not run on hardware
//
#include <hip/hip_runtime.h>
#include <float.h>

#define N_NODES 30000
#define N_EDGES 400000
#define HID 128
#define NH 4
#define CH 128
#define HC 512       // NH*CH
#define NLAYER 3
#define NGRAPH 64
#define ED 3
#define EF_TOT (N_EDGES + N_NODES)

typedef __attribute__((ext_vector_type(16))) __bf16 v16bf;
typedef __attribute__((ext_vector_type(8)))  float  v8f;

__device__ __forceinline__ unsigned short f2bf(float f) {
  unsigned int u = __float_as_uint(f);
  unsigned int r = u + 0x7FFFu + ((u >> 16) & 1u);  // round-to-nearest-even
  return (unsigned short)(r >> 16);
}

__device__ __forceinline__ void atomAddF(float* p, float v) {
  unsafeAtomicAdd(p, v);   // hardware global_atomic_add_f32
}

// float atomic max via signed/unsigned integer ordering trick
__device__ __forceinline__ void atomicMaxF(float* addr, float v) {
  if (v >= 0.f) atomicMax((int*)addr, __float_as_int(v));
  else          atomicMin((unsigned int*)addr, __float_as_uint(v));
}

// ---------------------------------------------------------------- utilities
__global__ void k_fill(float* p, int n, float v) {
  int t = blockIdx.x * blockDim.x + threadIdx.x;
  if (t < n) p[t] = v;
}

__global__ void k_cvt_bf16(const float* __restrict__ src,
                           unsigned short* __restrict__ dst, int n) {
  int t = blockIdx.x * blockDim.x + threadIdx.x;
  if (t < n) dst[t] = f2bf(src[t]);
}

// ------------------------------------------------- self-loop edge_attr mean
__global__ void k_deg_ea(const int* __restrict__ dst, const float* __restrict__ ea,
                         float* __restrict__ deg, float* __restrict__ easum) {
  int e = blockIdx.x * blockDim.x + threadIdx.x;
  if (e >= N_EDGES) return;
  int d = dst[e];
  atomAddF(&deg[d], 1.f);
  atomAddF(&easum[d * ED + 0], ea[e * ED + 0]);
  atomAddF(&easum[d * ED + 1], ea[e * ED + 1]);
  atomAddF(&easum[d * ED + 2], ea[e * ED + 2]);
}

__global__ void k_ea_mean(const float* __restrict__ deg, float* __restrict__ easum) {
  int t = blockIdx.x * blockDim.x + threadIdx.x;
  if (t >= N_NODES * ED) return;
  float d = deg[t / ED];
  easum[t] /= (d < 1.f ? 1.f : d);
}

// --------------------------------------------- pack W into WMMA B-fragments
// Wp[layer][kstep(4)][coltile(32)][lane(32)][elem(16)]  (bf16)
// lane<16  -> column = coltile*16+lane,    elem j -> K = kstep*32 + j
// lane>=16 -> column = coltile*16+lane-16, elem j -> K = kstep*32 + 16 + j
__global__ void k_packW(const float* __restrict__ W, unsigned short* __restrict__ Wp) {
  int t = blockIdx.x * blockDim.x + threadIdx.x;     // one 16-elem group each
  if (t >= NLAYER * 4 * 32 * 32) return;
  int layer = t / (4 * 32 * 32);
  int r = t % (4 * 32 * 32);
  int kstep = r / (32 * 32);
  int coltile = (r / 32) % 32;
  int lane = r % 32;
  int col = coltile * 16 + (lane & 15);
  int kb = (lane >> 4) * 16;
  const float* Wl = W + (size_t)layer * HID * HC;
  unsigned short* dst = Wp + (size_t)t * 16;
#pragma unroll
  for (int j = 0; j < 16; ++j)
    dst[j] = f2bf(Wl[(size_t)(kstep * 32 + kb + j) * HC + col]);
}

// ------------------------------------------------------- WMMA bf16 GEMM
// C[nrows,512] = A[nrows,128](bf16) @ W(packed bf16), f32 accumulate.
// One wave computes two 16x16 tiles sharing its A fragment. K=128 -> 4 steps.
__global__ __launch_bounds__(256)
void k_gemm_bf16_wmma(const unsigned short* __restrict__ A,
                      const unsigned short* __restrict__ Bp,
                      float* __restrict__ C, int nrows) {
  const int K = HID, NC = HC;
  int wave = threadIdx.x >> 5;                 // 0..7
  int lane = threadIdx.x & 31;
  int row0 = blockIdx.x * 16;
  int ct0 = blockIdx.y * 16 + wave * 2;        // this wave: col tiles ct0, ct0+1
  if (row0 >= nrows) return;

  int mrow = row0 + (lane & 15);
  int khalf = (lane >> 4);                     // A: K-half select (0/1)
  int col = (lane & 15);

  union F { v16bf v; unsigned int u[8]; uint4 q[2]; };
  v8f acc0 = {0.f, 0.f, 0.f, 0.f, 0.f, 0.f, 0.f, 0.f};
  v8f acc1 = acc0;

  const unsigned short* arow = A + (size_t)mrow * K;
#pragma unroll
  for (int kstep = 0; kstep < 4; ++kstep) {
    F a, b0, b1;
    // A 16x32 bf16 fragment: two contiguous 16B loads
    a.q[0] = *(const uint4*)(arow + kstep * 32 + khalf * 16);
    a.q[1] = *(const uint4*)(arow + kstep * 32 + khalf * 16 + 8);
    // B fragments: packed, fully coalesced 16B loads
    const unsigned short* bbase0 =
        Bp + ((size_t)(kstep * 32 + ct0) * 32 + lane) * 16;
    const unsigned short* bbase1 =
        Bp + ((size_t)(kstep * 32 + ct0 + 1) * 32 + lane) * 16;
    b0.q[0] = *(const uint4*)(bbase0);
    b0.q[1] = *(const uint4*)(bbase0 + 8);
    b1.q[0] = *(const uint4*)(bbase1);
    b1.q[1] = *(const uint4*)(bbase1 + 8);
    acc0 = __builtin_amdgcn_wmma_f32_16x16x32_bf16(false, a.v, false, b0.v,
                                                   (short)0, acc0, false, false);
    acc1 = __builtin_amdgcn_wmma_f32_16x16x32_bf16(false, a.v, false, b1.v,
                                                   (short)0, acc1, false, false);
  }
  // C/D layout: lane<16 rows 0..7, lane>=16 rows 8..15
#pragma unroll
  for (int r = 0; r < 8; ++r) {
    int m = row0 + r + (khalf ? 8 : 0);
    C[(size_t)m * NC + ct0 * 16 + col] = acc0[r];
    C[(size_t)m * NC + (ct0 + 1) * 16 + col] = acc1[r];
  }
}

// ----------------------------------------------------- attention node scores
__global__ void k_scores(const float* __restrict__ xh, const float* __restrict__ asrc,
                         const float* __restrict__ adst, float* __restrict__ a_s,
                         float* __restrict__ a_d) {
  int t = blockIdx.x * blockDim.x + threadIdx.x;
  if (t >= N_NODES * NH) return;
  int n = t >> 2, h = t & 3;
  const float4* xp = (const float4*)(xh + (size_t)n * HC + h * CH);
  const float4* s1 = (const float4*)(asrc + h * CH);
  const float4* s2 = (const float4*)(adst + h * CH);
  float accs = 0.f, accd = 0.f;
  for (int c = 0; c < CH / 4; ++c) {
    float4 v = xp[c], w1 = s1[c], w2 = s2[c];
    accs += v.x * w1.x + v.y * w1.y + v.z * w1.z + v.w * w1.w;
    accd += v.x * w2.x + v.y * w2.y + v.z * w2.z + v.w * w2.w;
  }
  a_s[t] = accs; a_d[t] = accd;
}

// pre-contract We with att_e: watt[d][h] = sum_c We[d, h*CH+c] * att_e[h,c]
__global__ void k_watt(const float* __restrict__ We, const float* __restrict__ att_e,
                       float* __restrict__ watt) {
  int t = threadIdx.x;
  if (t >= ED * NH) return;
  int d = t / NH, h = t % NH;
  float acc = 0.f;
  for (int c = 0; c < CH; ++c) acc += We[d * HC + h * CH + c] * att_e[h * CH + c];
  watt[d * NH + h] = acc;
}

// ----------------------------------------------------- edge logits + seg max
__global__ void k_logits(const int* __restrict__ src, const int* __restrict__ dst,
                         const float* __restrict__ ea, const float* __restrict__ eamean,
                         const float* __restrict__ watt, const float* __restrict__ a_s,
                         const float* __restrict__ a_d, float* __restrict__ logit,
                         float* __restrict__ mx) {
  int e = blockIdx.x * blockDim.x + threadIdx.x;
  if (e >= EF_TOT) return;
  int s, d; const float* a3;
  if (e < N_EDGES) { s = src[e]; d = dst[e]; a3 = ea + (size_t)e * ED; }
  else             { s = d = e - N_EDGES;    a3 = eamean + (size_t)(e - N_EDGES) * ED; }
  float e0 = a3[0], e1 = a3[1], e2 = a3[2];
#pragma unroll
  for (int h = 0; h < NH; ++h) {
    float l = a_s[s * NH + h] + a_d[d * NH + h]
            + e0 * watt[h] + e1 * watt[NH + h] + e2 * watt[2 * NH + h];
    l = l > 0.f ? l : 0.2f * l;                 // leaky relu
    logit[(size_t)e * NH + h] = l;
    atomicMaxF(&mx[d * NH + h], l);
  }
}

__global__ void k_expden(const int* __restrict__ dst, const float* __restrict__ logit,
                         const float* __restrict__ mx, float* __restrict__ ex,
                         float* __restrict__ den) {
  int e = blockIdx.x * blockDim.x + threadIdx.x;
  if (e >= EF_TOT) return;
  int d = (e < N_EDGES) ? dst[e] : e - N_EDGES;
#pragma unroll
  for (int h = 0; h < NH; ++h) {
    float v = __expf(logit[(size_t)e * NH + h] - mx[d * NH + h]);
    ex[(size_t)e * NH + h] = v;
    atomAddF(&den[d * NH + h], v);
  }
}

// ---------------------------------------------- weighted message scatter-add
// one block per edge; thread t -> head t>>5, channels 4*(t&31)..+3 (float4 gather)
__global__ __launch_bounds__(128)
void k_message(const int* __restrict__ src, const int* __restrict__ dst,
               const float* __restrict__ xh, const float* __restrict__ ex,
               const float* __restrict__ den, float* __restrict__ outacc) {
  int e = blockIdx.x;
  if (e >= EF_TOT) return;
  int s, d;
  if (e < N_EDGES) { s = src[e]; d = dst[e]; }
  else             { s = d = e - N_EDGES; }
  int tid = threadIdx.x;
  __shared__ float coef[NH];
  if (tid < NH) coef[tid] = ex[(size_t)e * NH + tid] / den[d * NH + tid];
  __syncthreads();
  int h = tid >> 5;
  int c4 = (tid & 31) * 4;
  float w = coef[h];
  const float4 v = *(const float4*)(xh + (size_t)s * HC + h * CH + c4);
  float* od = outacc + (size_t)d * HC + h * CH + c4;
  atomAddF(od + 0, v.x * w);
  atomAddF(od + 1, v.y * w);
  atomAddF(od + 2, v.z * w);
  atomAddF(od + 3, v.w * w);
}

// ------------------------------------------------- head mean + bias -> h
__global__ void k_headmean(const float* __restrict__ outacc, const float* __restrict__ bias,
                           float* __restrict__ h) {
  int t = blockIdx.x * blockDim.x + threadIdx.x;
  if (t >= N_NODES * HID) return;
  int n = t >> 7, c = t & 127;
  const float* o = outacc + (size_t)n * HC;
  h[t] = 0.25f * (o[c] + o[CH + c] + o[2 * CH + c] + o[3 * CH + c]) + bias[c];
}

// ------------------------------ BN stats: one block per channel, no atomics
__global__ __launch_bounds__(256)
void k_bnstats(const float* __restrict__ h, float* __restrict__ sums,
               float* __restrict__ sumsq) {
  int c = blockIdx.x;                 // 0..127
  float s = 0.f, q = 0.f;
  for (int n = threadIdx.x; n < N_NODES; n += 256) {
    float v = h[(size_t)n * HID + c];
    s += v; q += v * v;
  }
  __shared__ float sh[256], shq[256];
  sh[threadIdx.x] = s; shq[threadIdx.x] = q;
  __syncthreads();
  for (int stride = 128; stride > 0; stride >>= 1) {
    if (threadIdx.x < stride) {
      sh[threadIdx.x] += sh[threadIdx.x + stride];
      shq[threadIdx.x] += shq[threadIdx.x + stride];
    }
    __syncthreads();
  }
  if (threadIdx.x == 0) { sums[c] = sh[0]; sumsq[c] = shq[0]; }
}

__global__ void k_bnrelu(float* __restrict__ h, unsigned short* __restrict__ hb,
                         const float* __restrict__ sums, const float* __restrict__ sumsq,
                         const float* __restrict__ gamma, const float* __restrict__ beta) {
  int t = blockIdx.x * blockDim.x + threadIdx.x;
  if (t >= N_NODES * HID) return;
  int c = t & 127;
  const float inv = 1.f / (float)N_NODES;
  float mu = sums[c] * inv;
  float var = sumsq[c] * inv - mu * mu;
  float y = (h[t] - mu) * rsqrtf(var + 1e-5f) * gamma[c] + beta[c];
  y = y > 0.f ? y : 0.f;
  h[t] = y;
  hb[t] = f2bf(y);
}

// --------------------------------------------------------------- dense MLPs
__global__ void k_mlp(const float* __restrict__ in, const float* __restrict__ Wm,
                      const float* __restrict__ bv, float* __restrict__ out,
                      int nrows, int K, int J, int doRelu) {
  int t = blockIdx.x * blockDim.x + threadIdx.x;
  if (t >= nrows * J) return;
  int n = t / J, j = t % J;
  const float* ip = in + (size_t)n * K;
  float acc = bv[j];
  for (int k = 0; k < K; ++k) acc += ip[k] * Wm[(size_t)k * J + j];
  out[(size_t)n * J + j] = (doRelu && acc < 0.f) ? 0.f : acc;
}

// ------------------------------------------------------------- graph pooling
__global__ __launch_bounds__(128)
void k_pool(const float* __restrict__ h, const int* __restrict__ batch,
            float* __restrict__ gsum, float* __restrict__ gmax, float* __restrict__ cnt) {
  int n = blockIdx.x;
  if (n >= N_NODES) return;
  int c = threadIdx.x;
  int b = batch[n];
  float v = h[(size_t)n * HID + c];
  atomAddF(&gsum[b * HID + c], v);
  atomicMaxF(&gmax[b * HID + c], v);
  if (c == 0) atomAddF(&cnt[b], 1.f);
}

__global__ void k_grcat(const float* __restrict__ gsum, const float* __restrict__ gmax,
                        const float* __restrict__ cnt, float* __restrict__ gr) {
  int t = blockIdx.x * blockDim.x + threadIdx.x;
  if (t >= NGRAPH * HID) return;
  int g = t >> 7, c = t & 127;
  float cn = cnt[g]; cn = cn < 1.f ? 1.f : cn;
  gr[(size_t)g * 256 + c]        = gsum[t] / cn;
  gr[(size_t)g * 256 + HID + c]  = gmax[t];
}

// ================================================================== launch
extern "C" void kernel_launch(void* const* d_in, const int* in_sizes, int n_in,
                              void* d_out, int out_size, void* d_ws, size_t ws_size,
                              hipStream_t stream) {
  const float* x        = (const float*)d_in[0];
  const int*   ei       = (const int*)d_in[1];
  const float* ea       = (const float*)d_in[2];
  const int*   batch    = (const int*)d_in[3];
  const float* W        = (const float*)d_in[4];
  const float* att_src  = (const float*)d_in[5];
  const float* att_dst  = (const float*)d_in[6];
  const float* We       = (const float*)d_in[7];
  const float* att_e    = (const float*)d_in[8];
  const float* bias     = (const float*)d_in[9];
  const float* gamma    = (const float*)d_in[10];
  const float* beta     = (const float*)d_in[11];
  const float* nw1 = (const float*)d_in[12]; const float* nb1 = (const float*)d_in[13];
  const float* nw2 = (const float*)d_in[14]; const float* nb2 = (const float*)d_in[15];
  const float* nw3 = (const float*)d_in[16]; const float* nb3 = (const float*)d_in[17];
  const float* gw1 = (const float*)d_in[18]; const float* gb1 = (const float*)d_in[19];
  const float* gw2 = (const float*)d_in[20]; const float* gb2 = (const float*)d_in[21];
  const float* gw3 = (const float*)d_in[22]; const float* gb3 = (const float*)d_in[23];
  float* out = (float*)d_out;

  const int* src = ei;
  const int* dst = ei + N_EDGES;

  // -------- workspace carve (256B aligned) --------
  char* base = (char*)d_ws;
  auto carve = [&](size_t bytes) -> void* {
    void* p = (void*)base;
    base += (bytes + 255) & ~(size_t)255;
    return p;
  };
  float*          deg    = (float*)carve((size_t)N_NODES * 4);
  float*          eamean = (float*)carve((size_t)N_NODES * ED * 4);
  unsigned short* Wp     = (unsigned short*)carve((size_t)NLAYER * HID * HC * 2);
  unsigned short* hb     = (unsigned short*)carve((size_t)N_NODES * HID * 2);
  float*          h      = (float*)carve((size_t)N_NODES * HID * 4);
  float*          xh     = (float*)carve((size_t)N_NODES * HC * 4);
  float*          outacc = (float*)carve((size_t)N_NODES * HC * 4);
  float*          a_s    = (float*)carve((size_t)N_NODES * NH * 4);
  float*          a_d    = (float*)carve((size_t)N_NODES * NH * 4);
  float*          logit  = (float*)carve((size_t)EF_TOT * NH * 4);
  float*          ex     = (float*)carve((size_t)EF_TOT * NH * 4);
  float*          mx     = (float*)carve((size_t)N_NODES * NH * 4);
  float*          den    = (float*)carve((size_t)N_NODES * NH * 4);
  float*          sums   = (float*)carve(HID * 4);
  float*          sumsq  = (float*)carve(HID * 4);
  float*          watt   = (float*)carve(ED * NH * 4);
  float*          t1     = (float*)carve((size_t)N_NODES * 64 * 4);
  float*          t2     = (float*)carve((size_t)N_NODES * 32 * 4);
  float*          gsum   = (float*)carve((size_t)NGRAPH * HID * 4);
  float*          gmax   = (float*)carve((size_t)NGRAPH * HID * 4);
  float*          cnt    = (float*)carve((size_t)NGRAPH * 4);
  float*          gr     = (float*)carve((size_t)NGRAPH * 256 * 4);
  float*          g1     = (float*)carve((size_t)NGRAPH * HID * 4);
  float*          g2     = (float*)carve((size_t)NGRAPH * 64 * 4);

  auto nb = [](int n, int b) { return (n + b - 1) / b; };

  // -------- self-loop edge_attr mean --------
  k_fill<<<nb(N_NODES, 256), 256, 0, stream>>>(deg, N_NODES, 0.f);
  k_fill<<<nb(N_NODES * ED, 256), 256, 0, stream>>>(eamean, N_NODES * ED, 0.f);
  k_deg_ea<<<nb(N_EDGES, 256), 256, 0, stream>>>(dst, ea, deg, eamean);
  k_ea_mean<<<nb(N_NODES * ED, 256), 256, 0, stream>>>(deg, eamean);

  // -------- pack weights to WMMA fragment layout; features to bf16 --------
  k_packW<<<nb(NLAYER * 4 * 32 * 32, 256), 256, 0, stream>>>(W, Wp);
  k_cvt_bf16<<<nb(N_NODES * HID, 256), 256, 0, stream>>>(x, hb, N_NODES * HID);

  for (int i = 0; i < NLAYER; ++i) {
    const unsigned short* Wb  = Wp + (size_t)i * HID * HC;
    const float* asv = att_src + (size_t)i * NH * CH;
    const float* adv = att_dst + (size_t)i * NH * CH;
    const float* Wev = We + (size_t)i * ED * HC;
    const float* aev = att_e + (size_t)i * NH * CH;
    const float* bsv = bias + (size_t)i * HID;
    const float* gmv = gamma + (size_t)i * HID;
    const float* btv = beta + (size_t)i * HID;

    // xh = h @ W[i]  (WMMA bf16, packed B, 2 tiles/wave)
    dim3 ggrid(N_NODES / 16, 2);
    k_gemm_bf16_wmma<<<ggrid, 256, 0, stream>>>(hb, Wb, xh, N_NODES);

    k_scores<<<nb(N_NODES * NH, 256), 256, 0, stream>>>(xh, asv, adv, a_s, a_d);
    k_watt<<<1, 16, 0, stream>>>(Wev, aev, watt);

    k_fill<<<nb(N_NODES * NH, 256), 256, 0, stream>>>(mx, N_NODES * NH, -3.402823466e38f);
    k_fill<<<nb(N_NODES * NH, 256), 256, 0, stream>>>(den, N_NODES * NH, 0.f);
    k_fill<<<nb(N_NODES * HC, 256), 256, 0, stream>>>(outacc, N_NODES * HC, 0.f);

    k_logits<<<nb(EF_TOT, 256), 256, 0, stream>>>(src, dst, ea, eamean, watt,
                                                  a_s, a_d, logit, mx);
    k_expden<<<nb(EF_TOT, 256), 256, 0, stream>>>(dst, logit, mx, ex, den);
    k_message<<<EF_TOT, 128, 0, stream>>>(src, dst, xh, ex, den, outacc);

    k_headmean<<<nb(N_NODES * HID, 256), 256, 0, stream>>>(outacc, bsv, h);
    k_bnstats<<<HID, 256, 0, stream>>>(h, sums, sumsq);
    k_bnrelu<<<nb(N_NODES * HID, 256), 256, 0, stream>>>(h, hb, sums, sumsq, gmv, btv);
  }

  // -------- node classifier MLP: 128 -> 64 -> 32 -> 2 --------
  k_mlp<<<nb(N_NODES * 64, 256), 256, 0, stream>>>(h, nw1, nb1, t1, N_NODES, 128, 64, 1);
  k_mlp<<<nb(N_NODES * 32, 256), 256, 0, stream>>>(t1, nw2, nb2, t2, N_NODES, 64, 32, 1);
  k_mlp<<<nb(N_NODES * 2, 256), 256, 0, stream>>>(t2, nw3, nb3, out, N_NODES, 32, 2, 0);

  // -------- graph pooling + classifier MLP: 256 -> 128 -> 64 -> 2 --------
  k_fill<<<nb(NGRAPH * HID, 256), 256, 0, stream>>>(gsum, NGRAPH * HID, 0.f);
  k_fill<<<nb(NGRAPH * HID, 256), 256, 0, stream>>>(gmax, NGRAPH * HID, -3.402823466e38f);
  k_fill<<<1, 64, 0, stream>>>(cnt, NGRAPH, 0.f);
  k_pool<<<N_NODES, 128, 0, stream>>>(h, batch, gsum, gmax, cnt);
  k_grcat<<<nb(NGRAPH * HID, 256), 256, 0, stream>>>(gsum, gmax, cnt, gr);

  k_mlp<<<nb(NGRAPH * 128, 256), 256, 0, stream>>>(gr, gw1, gb1, g1, NGRAPH, 256, 128, 1);
  k_mlp<<<nb(NGRAPH * 64, 256), 256, 0, stream>>>(g1, gw2, gb2, g2, NGRAPH, 128, 64, 1);
  k_mlp<<<1, 128, 0, stream>>>(g2, gw3, gb3, out + (size_t)N_NODES * 2, NGRAPH, 64, 2, 0);
}
